// SelfAttentionLocal_12103217840303
// MI455X (gfx1250) — matrix operand order repaired and checked by reading, not compiled, not run on hardware
//
#include <hip/hip_runtime.h>
#include <hip/hip_bf16.h>

// ---------------------------------------------------------------------------
// SelfAttentionLocal for MI455X (gfx1250, wave32, WMMA bf16 16x16x32).
// N=1024, DIM=512, HEADS=8, hd=64. All GEMMs on v_wmma_f32_16x16x32_bf16
// (fp32 accumulate). q/k/vT and softmax-P are staged once as bf16 so WMMA
// fragment loads are contiguous b128 loads with no inner-loop conversions.
// Intermediates (51MB) stay resident in the 192MB L2.
// ---------------------------------------------------------------------------

typedef __attribute__((ext_vector_type(16))) __bf16 v16bf;
typedef __attribute__((ext_vector_type(8)))  __bf16 v8bf;
typedef __attribute__((ext_vector_type(8)))  float  v8f;

#define NTOK   1024
#define DIM    512
#define QKVDIM 1536
#define HEADS  8
#define HDIM   64
#define SCALE  0.125f   // (512/8)^-0.5, exact power of two -> folded into q

#define WMMA_BF16(a, b, c) \
  __builtin_amdgcn_wmma_f32_16x16x32_bf16(false, (a), false, (b), (short)0, (c), false, false)

// ---- WMMA fragment loaders (layouts per CDNA5 ISA 7.12.2) ------
// A 16x32: lane g=lane/16, m=lane%16; elems 0..7 -> K=8g+0..7, 8..15 -> K=16+8g+0..7
// B 32x16: lane n=lane%16, g=lane/16; elems 0..7 -> K=8g+0..7, 8..15 -> K=16+8g+0..7
// C 16x16 f32: VGPR r, lane -> row r+8*(lane/16), col lane%16

__device__ __forceinline__ v16bf load_a_f32(const float* __restrict__ base,
                                            int ld, int row0, int k0) {
  int lane = threadIdx.x & 31;
  int g = lane >> 4, m = lane & 15;
  const float* p = base + (size_t)(row0 + m) * ld + k0;
  v16bf a;
#pragma unroll
  for (int e = 0; e < 8; ++e) a[e]     = (__bf16)p[8 * g + e];
#pragma unroll
  for (int e = 0; e < 8; ++e) a[8 + e] = (__bf16)p[16 + 8 * g + e];
  return a;
}

__device__ __forceinline__ v16bf load_b_f32_nt(const float* __restrict__ base,
                                               int ld, int n0, int k0) {
  int lane = threadIdx.x & 31;
  int g = lane >> 4, n = lane & 15;
  const float* p = base + (size_t)(n0 + n) * ld + k0;
  v16bf b;
#pragma unroll
  for (int e = 0; e < 8; ++e) { b[e] = (__bf16)p[8 * g + e]; b[8 + e] = (__bf16)p[16 + 8 * g + e]; }
  return b;
}

__device__ __forceinline__ v16bf load_a_bf16(const __bf16* __restrict__ base,
                                             int ld, int row0, int k0) {
  int lane = threadIdx.x & 31;
  int g = lane >> 4, m = lane & 15;
  const __bf16* p = base + (size_t)(row0 + m) * ld + k0;
  v8bf lo = *(const v8bf*)(p + 8 * g);        // 16B contiguous
  v8bf hi = *(const v8bf*)(p + 16 + 8 * g);   // 16B contiguous
  v16bf a;
#pragma unroll
  for (int e = 0; e < 8; ++e) { a[e] = lo[e]; a[8 + e] = hi[e]; }
  return a;
}

__device__ __forceinline__ v16bf load_b_bf16_nt(const __bf16* __restrict__ base,
                                                int ld, int n0, int k0) {
  int lane = threadIdx.x & 31;
  int g = lane >> 4, n = lane & 15;
  const __bf16* p = base + (size_t)(n0 + n) * ld + k0;
  v8bf lo = *(const v8bf*)(p + 8 * g);
  v8bf hi = *(const v8bf*)(p + 16 + 8 * g);
  v16bf b;
#pragma unroll
  for (int e = 0; e < 8; ++e) { b[e] = lo[e]; b[8 + e] = hi[e]; }
  return b;
}

__device__ __forceinline__ void store_c(float* __restrict__ base, int ld,
                                        int row0, int col0, v8f c) {
  int lane = threadIdx.x & 31;
  int g = lane >> 4, n = lane & 15;
#pragma unroll
  for (int r = 0; r < 8; ++r)
    base[(size_t)(row0 + r + 8 * g) * ld + col0 + n] = c[r];
}

// ---- Kernel 1: QKV projection + bf16 staging ------------------------------
// C[1024][1536] = x @ Wqkv^T, scattered to qs[h][n][d] (*SCALE), ks[h][n][d],
// vT[h][d][n], all bf16. One wave per 16x64 strip (A reused over 4 B tiles).
__global__ void k_qkv_gemm(const float* __restrict__ x,
                           const float* __restrict__ Wqkv,
                           __bf16* __restrict__ qs, __bf16* __restrict__ ks,
                           __bf16* __restrict__ vT) {
  int wave = blockIdx.x * 4 + (threadIdx.x >> 5);   // 64 * 24 = 1536 waves
  int mt = wave / 24, ng = wave % 24;
  int row0 = mt * 16, colbase = ng * 64;
  v8f acc[4];
#pragma unroll
  for (int t = 0; t < 4; ++t) acc[t] = (v8f){0.f,0.f,0.f,0.f,0.f,0.f,0.f,0.f};
  for (int k0 = 0; k0 < DIM; k0 += 32) {
    v16bf a = load_a_f32(x, DIM, row0, k0);
#pragma unroll
    for (int t = 0; t < 4; ++t) {
      v16bf b = load_b_f32_nt(Wqkv, DIM, colbase + t * 16, k0);
      acc[t] = WMMA_BF16(a, b, acc[t]);
    }
  }
  int lane = threadIdx.x & 31;
  int g = lane >> 4, n = lane & 15;
  // whole 64-col group lies in one section (q/k/v) and one head
  int sec = colbase >> 9;            // 0=q 1=k 2=v
  int h   = (colbase & 511) >> 6;
#pragma unroll
  for (int t = 0; t < 4; ++t) {
    int d0 = (colbase & 63) + t * 16 + n;   // d within head (colbase%64==0 here)
#pragma unroll
    for (int r = 0; r < 8; ++r) {
      int row = row0 + r + 8 * g;
      float val = acc[t][r];
      if (sec == 0)
        qs[((size_t)h * NTOK + row) * HDIM + d0] = (__bf16)(val * SCALE);
      else if (sec == 1)
        ks[((size_t)h * NTOK + row) * HDIM + d0] = (__bf16)val;
      else
        vT[((size_t)h * HDIM + d0) * NTOK + row] = (__bf16)val;
    }
  }
}

// ---- Kernel 2: location bias  ATT[h][n][m] = log(relu(emb.Wloc[h]+b)+1e-6)
__global__ void k_loc_bias(const float* __restrict__ locs,
                           const float* __restrict__ Wloc,
                           const float* __restrict__ bloc,
                           float* __restrict__ ATT) {
  // 100 / 1000^(t/8), t = 0..7
  const float inv_dim[8] = {100.0f,      42.169652f, 17.782795f, 7.4989421f,
                            3.1622777f,  1.3335215f, 0.56234133f, 0.23713737f};
  int idx = blockIdx.x * blockDim.x + threadIdx.x;  // n*1024 + m
  int n = idx >> 10, m = idx & 1023;

  float nx1 = locs[n * 4 + 0], ny1 = locs[n * 4 + 1];
  float nx2 = locs[n * 4 + 2], ny2 = locs[n * 4 + 3];
  float mx1 = locs[m * 4 + 0], my1 = locs[m * 4 + 1];
  float mx2 = locs[m * 4 + 2], my2 = locs[m * 4 + 3];

  float wn = nx2 - nx1 + 1.f, hn = ny2 - ny1 + 1.f;
  float wm = mx2 - mx1 + 1.f, hm = my2 - my1 + 1.f;
  float cxn = 0.5f * (nx1 + nx2), cyn = 0.5f * (ny1 + ny2);
  float cxm = 0.5f * (mx1 + mx2), cym = 0.5f * (my1 + my2);

  float pos[4];
  pos[0] = __logf(fmaxf(fabsf((cxn - cxm) / wn), 1e-3f));
  pos[1] = __logf(fmaxf(fabsf((cyn - cym) / hn), 1e-3f));
  pos[2] = __logf(wm / wn);
  pos[3] = __logf(hm / hn);

  float emb[64];
#pragma unroll
  for (int c = 0; c < 4; ++c) {
#pragma unroll
    for (int t = 0; t < 8; ++t) {
      float d = pos[c] * inv_dim[t];
      emb[c * 16 + t]     = __sinf(d);
      emb[c * 16 + 8 + t] = __cosf(d);
    }
  }
#pragma unroll
  for (int h = 0; h < HEADS; ++h) {
    float acc = bloc[h];
#pragma unroll
    for (int f = 0; f < 64; ++f) acc = fmaf(emb[f], Wloc[h * 64 + f], acc);
    acc = fmaxf(acc, 0.f);
    ATT[(size_t)h * NTOK * NTOK + idx] = __logf(acc + 1e-6f);
  }
}

// ---- Kernel 3: scores  ATT[h][i][j] += q_h[i].k_h[j]  (q prescaled) -------
// One wave per 16x64 strip: A fragments held in registers across 4 j-tiles.
__global__ void k_score(const __bf16* __restrict__ qs,
                        const __bf16* __restrict__ ks,
                        float* __restrict__ ATT) {
  int wave = blockIdx.x * 4 + (threadIdx.x >> 5);   // 8 * 64 * 16 = 8192
  int h  = wave >> 10;
  int it = (wave >> 4) & 63, jg = wave & 15;
  const __bf16* q = qs + (size_t)h * NTOK * HDIM;
  const __bf16* k = ks + (size_t)h * NTOK * HDIM;
  v16bf a0 = load_a_bf16(q, HDIM, it * 16, 0);
  v16bf a1 = load_a_bf16(q, HDIM, it * 16, 32);
  float* att = ATT + (size_t)h * NTOK * NTOK;
  int lane = threadIdx.x & 31;
  int g = lane >> 4, n = lane & 15;
#pragma unroll
  for (int t = 0; t < 4; ++t) {
    int j0 = jg * 64 + t * 16;
    v16bf b0 = load_b_bf16_nt(k, HDIM, j0, 0);
    v16bf b1 = load_b_bf16_nt(k, HDIM, j0, 32);
    v8f acc = {0.f,0.f,0.f,0.f,0.f,0.f,0.f,0.f};
    acc = WMMA_BF16(a0, b0, acc);
    acc = WMMA_BF16(a1, b1, acc);
#pragma unroll
    for (int r = 0; r < 8; ++r) {
      size_t off = (size_t)(it * 16 + r + 8 * g) * NTOK + j0 + n;
      att[off] = acc[r] + att[off];     // bias already stored in ATT
    }
  }
}

// ---- Kernel 4: row softmax, fp32 scores -> bf16 probabilities -------------
__global__ void k_softmax(const float* __restrict__ ATT, __bf16* __restrict__ PB) {
  __shared__ float red_max[8];
  __shared__ float red_sum[8];
  const float* p = ATT + (size_t)blockIdx.x * NTOK;
  __bf16* pb     = PB  + (size_t)blockIdx.x * NTOK;
  int tid = threadIdx.x;

  float vmax = -3.0e38f;
#pragma unroll
  for (int t = 0; t < 4; ++t) vmax = fmaxf(vmax, p[tid + t * 256]);
#pragma unroll
  for (int s = 16; s; s >>= 1) vmax = fmaxf(vmax, __shfl_xor(vmax, s, 32));
  if ((tid & 31) == 0) red_max[tid >> 5] = vmax;
  __syncthreads();
#pragma unroll
  for (int w = 0; w < 8; ++w) vmax = fmaxf(vmax, red_max[w]);

  float vals[4], sum = 0.f;
#pragma unroll
  for (int t = 0; t < 4; ++t) {
    vals[t] = __expf(p[tid + t * 256] - vmax);
    sum += vals[t];
  }
#pragma unroll
  for (int s = 16; s; s >>= 1) sum += __shfl_xor(sum, s, 32);
  if ((tid & 31) == 0) red_sum[tid >> 5] = sum;
  __syncthreads();
  sum = 0.f;
#pragma unroll
  for (int w = 0; w < 8; ++w) sum += red_sum[w];

  float inv = __frcp_rn(sum);
#pragma unroll
  for (int t = 0; t < 4; ++t) pb[tid + t * 256] = (__bf16)(vals[t] * inv);
}

// ---- Kernel 5: out[n][h*64+d] = sum_m P[h][n][m] * vT[h][d][m] ------------
// One wave per 16x64 output strip: 32 k-steps x (1 A load + 4 B + 4 wmma).
__global__ void k_pv_gemm(const __bf16* __restrict__ PB,
                          const __bf16* __restrict__ vT,
                          float* __restrict__ out) {
  int wave = blockIdx.x * 4 + (threadIdx.x >> 5);   // 8 * 64 = 512 waves
  int h = wave >> 6, it = wave & 63;
  const __bf16* P  = PB + (size_t)h * NTOK * NTOK;
  const __bf16* vt = vT + (size_t)h * HDIM * NTOK;
  v8f acc[4];
#pragma unroll
  for (int t = 0; t < 4; ++t) acc[t] = (v8f){0.f,0.f,0.f,0.f,0.f,0.f,0.f,0.f};
  for (int k0 = 0; k0 < NTOK; k0 += 32) {
    v16bf a = load_a_bf16(P, NTOK, it * 16, k0);
#pragma unroll
    for (int t = 0; t < 4; ++t) {
      v16bf b = load_b_bf16_nt(vt, NTOK, t * 16, k0);  // B(K=m,n=d)=vT[d][m]
      acc[t] = WMMA_BF16(a, b, acc[t]);
    }
  }
#pragma unroll
  for (int t = 0; t < 4; ++t)
    store_c(out + h * HDIM, DIM, it * 16, t * 16, acc[t]);
}

// ---------------------------------------------------------------------------
extern "C" void kernel_launch(void* const* d_in, const int* in_sizes, int n_in,
                              void* d_out, int out_size, void* d_ws, size_t ws_size,
                              hipStream_t stream) {
  const float* x    = (const float*)d_in[0];
  // d_in[1] = x_reg: unused by the reference computation
  const float* locs = (const float*)d_in[2];
  const float* Wqkv = (const float*)d_in[3];
  const float* Wloc = (const float*)d_in[4];
  const float* bloc = (const float*)d_in[5];
  float* out = (float*)d_out;

  // Workspace layout (bytes):
  //   qs  bf16 [8][1024][64]    @ 0        (1MB)
  //   ks  bf16 [8][1024][64]    @ 1MB      (1MB)
  //   vT  bf16 [8][64][1024]    @ 2MB      (1MB)
  //   ATT f32  [8][1024][1024]  @ 3MB      (32MB)
  //   PB  bf16 [8][1024][1024]  @ 35MB     (16MB)
  char* ws = (char*)d_ws;
  __bf16* qs  = (__bf16*)(ws);
  __bf16* ks  = (__bf16*)(ws + (1u << 20));
  __bf16* vT  = (__bf16*)(ws + (2u << 20));
  float*  ATT = (float*) (ws + (3u << 20));
  __bf16* PB  = (__bf16*)(ws + (35u << 20));

  // 1) QKV projection + bf16 staging: 1536 waves, 4 waves/block
  k_qkv_gemm<<<1536 / 4, 128, 0, stream>>>(x, Wqkv, qs, ks, vT);
  // 2) geometry bias into ATT: 1M pairs
  k_loc_bias<<<(NTOK * NTOK) / 256, 256, 0, stream>>>(locs, Wloc, bloc, ATT);
  // 3) attention scores (+bias, in place): 8192 waves
  k_score<<<8192 / 4, 128, 0, stream>>>(qs, ks, ATT);
  // 4) softmax rows -> bf16 P: 8192 rows
  k_softmax<<<HEADS * NTOK, 256, 0, stream>>>(ATT, PB);
  // 5) P @ V: 512 waves
  k_pv_gemm<<<512 / 4, 128, 0, stream>>>(PB, vT, out);
}
